// MultiHeadAttention_21440476742467
// MI455X (gfx1250) — compile-verified, matrix operand order checked by reading, not run on hardware
//
#include <hip/hip_runtime.h>
#include <hip/hip_bf16.h>

// MHA forward for MI455X (gfx1250): bf16 WMMA GEMMs with async-to-LDS double
// buffering + register-pipelined flash-attention.
// DIM=1024, H=16, hd=64, B=2, T=2048.

#define DIMC   1024
#define NHEAD  16
#define HDIM   64
#define BATCH  2
#define SEQ    2048
#define ROWS   (BATCH * SEQ)          // 4096

typedef __bf16 v16bf __attribute__((ext_vector_type(16)));
typedef float  v8f   __attribute__((ext_vector_type(8)));
typedef unsigned int v4u __attribute__((ext_vector_type(4)));
typedef float  v4f   __attribute__((ext_vector_type(4)));
typedef __bf16 v4bf  __attribute__((ext_vector_type(4)));

union FragBF {
    v16bf f;
    v4u   q[2];
};

// ---- WMMA fragment gathers -------------------------------------------------
// A-fragment 16x32 bf16: lane L<16 -> row L, k 0..7 & 16..23; lane L>=16 ->
// row L-16, k 8..15 & 24..31.  Identical striping serves B (lane = column).
__device__ __forceinline__ v16bf load_frag_rows(const __bf16* base, int ld) {
    int lane = threadIdx.x & 31;
    int r    = lane & 15;
    int hi8  = (lane >> 4) << 3;
    const __bf16* p = base + r * ld + hi8;
    FragBF u;
    u.q[0] = *(const v4u*)(p);
    u.q[1] = *(const v4u*)(p + 16);
    return u.f;
}

// LDS tile fragment load: rows of 32 bf16 (64B) with 16B-chunk XOR swizzle
// (chunk' = chunk ^ (row & 3)) to spread banks across the 16-lane gather.
__device__ __forceinline__ v16bf load_frag_lds_sw(const __bf16* tile, int row0) {
    int lane = threadIdx.x & 31;
    int r    = row0 + (lane & 15);
    int c0   = (lane >> 4);          // chunk 0 or 1  (k 0..7 / 8..15)
    int c1   = c0 + 2;               // chunk 2 or 3  (k 16..23 / 24..31)
    int sw   = r & 3;
    FragBF u;
    u.q[0] = *(const v4u*)(tile + r * 32 + ((c0 ^ sw) << 3));
    u.q[1] = *(const v4u*)(tile + r * 32 + ((c1 ^ sw) << 3));
    return u.f;
}

__device__ __forceinline__ v8f wmma_bf16(v16bf a, v16bf b, v8f c) {
    return __builtin_amdgcn_wmma_f32_16x16x32_bf16(
        false, a, false, b, (short)0, c, false, false);
}

// ---- gfx1250 async global->LDS copy (ASYNCcnt-tracked) ---------------------
// One instruction moves 16B per lane (512B per wave) into LDS.
__device__ __forceinline__ void async_b128(unsigned lds_off, int gbyte_off,
                                           const __bf16* base) {
    asm volatile("global_load_async_to_lds_b128 %0, %1, %2"
                 :: "v"(lds_off), "v"(gbyte_off), "s"(base)
                 : "memory");
}

// ---------------------------------------------------------------- convert
__global__ void cvt_f32_bf16_kernel(const v4f* __restrict__ src,
                                    v4bf* __restrict__ dst, int n4) {
    int i = blockIdx.x * blockDim.x + threadIdx.x;
    if (i < n4) {
        v4f x = src[i];
        v4bf y;
#pragma unroll
        for (int e = 0; e < 4; ++e) y[e] = (__bf16)x[e];
        dst[i] = y;
    }
}

// ---------------------------------------------------------------- GEMM
// C[M,N] = A[M,K] * W[N,K]^T + bias[N];  M=4096, N=K=1024, bf16 in, f32 acc.
// 128-thread workgroup (4 waves), macro-tile 64(M) x 128(N), K-step 32.
// Double-buffered LDS tiles fed by global_load_async_to_lds_b128.
#define GMODE_QK  0   // store bf16 at [b][h][t][d]
#define GMODE_VT  1   // store bf16 at [b][h][d][t]   (transposed per head)
#define GMODE_F32 2   // store f32 row-major (final output)

#define BUF_ELEMS 6144u   // A 64x32 (2048) + B 128x32 (4096) bf16 per buffer

// Stage one K-slab into LDS buffer at byte offset `ldsbuf`.
// Chunk layout: A tile 256 x 16B chunks (2/thread), B tile 512 (4/thread).
__device__ __forceinline__ void issue_stage(const __bf16* A, const __bf16* W,
                                            int m0, int n0, int t, int kb,
                                            unsigned ldsbuf) {
#pragma unroll
    for (int q = 0; q < 2; ++q) {
        int ci = t + q * 128;
        int r  = ci >> 2, c = ci & 3;
        unsigned lofs = ldsbuf + (unsigned)((r << 6) + ((c ^ (r & 3)) << 4));
        int gofs = ((m0 + r) * DIMC + kb + (c << 3)) * 2;
        async_b128(lofs, gofs, A);
    }
#pragma unroll
    for (int q = 0; q < 4; ++q) {
        int ci = t + q * 128;
        int r  = ci >> 2, c = ci & 3;
        unsigned lofs = ldsbuf + 4096u + (unsigned)((r << 6) + ((c ^ (r & 3)) << 4));
        int gofs = ((n0 + r) * DIMC + kb + (c << 3)) * 2;
        async_b128(lofs, gofs, W);
    }
}

template <int MODE>
__global__ __launch_bounds__(128, 1)
void gemm_bf16_kernel(const __bf16* __restrict__ A,
                      const __bf16* __restrict__ W,
                      const float* __restrict__ bias,
                      void* __restrict__ out) {
    __shared__ __bf16 lds[2 * BUF_ELEMS];          // 24 KB
    const int t    = threadIdx.x;                  // 0..127
    const int lane = t & 31;
    const int wv   = t >> 5;                       // wave id 0..3
    const int n0   = blockIdx.x * 128;
    const int m0   = blockIdx.y * 64;
    const int cl   = lane & 15;
    const int hi8  = (lane >> 4) << 3;
    const unsigned lbase = (unsigned)(unsigned long long)(void*)lds;

    v8f acc[4][2];
#pragma unroll
    for (int i = 0; i < 4; ++i)
#pragma unroll
        for (int j = 0; j < 2; ++j) acc[i][j] = (v8f){};

    issue_stage(A, W, m0, n0, t, 0, lbase);

    const int NK = DIMC / 32;                      // 32 K-steps
    for (int k = 0; k < NK; ++k) {
        if (k + 1 < NK) {
            issue_stage(A, W, m0, n0, t, (k + 1) * 32,
                        lbase + ((unsigned)((k + 1) & 1)) * (BUF_ELEMS * 2u));
            asm volatile("s_wait_asynccnt 0x6" ::: "memory");  // stage k done
        } else {
            asm volatile("s_wait_asynccnt 0x0" ::: "memory");
        }
        __syncthreads();                            // stage k visible to WG

        const __bf16* As = lds + (k & 1) * BUF_ELEMS;
        const __bf16* Bs = As + 2048;
        v16bf a0 = load_frag_lds_sw(As,  0);
        v16bf a1 = load_frag_lds_sw(As, 16);
        v16bf a2 = load_frag_lds_sw(As, 32);
        v16bf a3 = load_frag_lds_sw(As, 48);
        v16bf b0 = load_frag_lds_sw(Bs, wv * 32);
        v16bf b1 = load_frag_lds_sw(Bs, wv * 32 + 16);
        acc[0][0] = wmma_bf16(a0, b0, acc[0][0]);
        acc[0][1] = wmma_bf16(a0, b1, acc[0][1]);
        acc[1][0] = wmma_bf16(a1, b0, acc[1][0]);
        acc[1][1] = wmma_bf16(a1, b1, acc[1][1]);
        acc[2][0] = wmma_bf16(a2, b0, acc[2][0]);
        acc[2][1] = wmma_bf16(a2, b1, acc[2][1]);
        acc[3][0] = wmma_bf16(a3, b0, acc[3][0]);
        acc[3][1] = wmma_bf16(a3, b1, acc[3][1]);
        __syncthreads();                            // done reading buffer k&1
    }

#pragma unroll
    for (int j = 0; j < 2; ++j) {
        const int col = n0 + wv * 32 + j * 16 + cl;
        const float bv = bias[col];
#pragma unroll
        for (int i = 0; i < 4; ++i) {
#pragma unroll
            for (int e = 0; e < 8; ++e) {
                const int row = m0 + i * 16 + e + hi8;
                const float val = acc[i][j][e] + bv;
                if (MODE == GMODE_F32) {
                    ((float*)out)[row * DIMC + col] = val;
                } else {
                    const int bb = row >> 11;          // / SEQ
                    const int tt = row & (SEQ - 1);
                    const int hh = col >> 6;           // / HDIM
                    const int dd = col & (HDIM - 1);
                    __bf16* o = (__bf16*)out;
                    if (MODE == GMODE_QK)
                        o[((bb * NHEAD + hh) * SEQ + tt) * HDIM + dd] = (__bf16)val;
                    else   // GMODE_VT
                        o[((bb * NHEAD + hh) * HDIM + dd) * SEQ + tt] = (__bf16)val;
                }
            }
        }
    }
}

// ---------------------------------------------------------------- attention
// One wave handles 16 query rows of one (batch, head). Flash-attention over
// key blocks of 32 with register pipelining:
//   - V fragments issued at loop top (latency hidden by S-WMMA + softmax)
//   - K fragments double-buffered one block ahead.
// __launch_bounds__(32,1): single wave per block, full VGPR budget (the
// default occupancy heuristic spilled the prefetch fragments to scratch).
// Scale is 1/sqrt(DIM) = 1/32 (matches reference).
__global__ __launch_bounds__(32, 1)
void attention_kernel(const __bf16* __restrict__ Qb,
                      const __bf16* __restrict__ Kb,
                      const __bf16* __restrict__ Vt,
                      __bf16* __restrict__ Ob) {
    const int t0 = blockIdx.x * 16;
    const int bh = blockIdx.y;               // b*NHEAD + h
    const int b  = bh / NHEAD;
    const int h  = bh % NHEAD;

    const __bf16* Qh = Qb + (size_t)bh * SEQ * HDIM;
    const __bf16* Kh = Kb + (size_t)bh * SEQ * HDIM;
    const __bf16* Vh = Vt + (size_t)bh * HDIM * SEQ;

    const int lane = threadIdx.x & 31;
    const int cl   = lane & 15;
    const int hi8  = (lane >> 4) << 3;
    const float L2E = 1.44269504f;
    const float SCALE = 0.03125f;            // 1/sqrt(1024)

    __shared__ __bf16 Pl[16 * 32];           // P staging (C-layout -> A-layout)

    v16bf qf0 = load_frag_rows(Qh + t0 * HDIM +  0, HDIM);
    v16bf qf1 = load_frag_rows(Qh + t0 * HDIM + 32, HDIM);

    v8f o[4];
#pragma unroll
    for (int c = 0; c < 4; ++c) o[c] = (v8f){};
    float m_s[8], l_s[8];
#pragma unroll
    for (int e = 0; e < 8; ++e) { m_s[e] = -3.0e38f; l_s[e] = 0.0f; }

    const int kb_end = t0 + 15;              // last key position needed

    // Preload K fragments for the first key block.
    v16bf kc[4];
    kc[0] = load_frag_rows(Kh +  0 * HDIM +  0, HDIM);
    kc[1] = load_frag_rows(Kh +  0 * HDIM + 32, HDIM);
    kc[2] = load_frag_rows(Kh + 16 * HDIM +  0, HDIM);
    kc[3] = load_frag_rows(Kh + 16 * HDIM + 32, HDIM);

    for (int kb = 0; kb <= kb_end; kb += 32) {
        // Issue V fragment loads early; consumed only after softmax.
        v16bf vf[4];
#pragma unroll
        for (int c = 0; c < 4; ++c)
            vf[c] = load_frag_rows(Vh + (c * 16) * SEQ + kb, SEQ);

        // S = Q K^T for 16 rows x 32 keys (two 16x16 tiles).
        v8f s0 = (v8f){}, s1 = (v8f){};
        s0 = wmma_bf16(qf0, kc[0], s0);
        s0 = wmma_bf16(qf1, kc[1], s0);
        s1 = wmma_bf16(qf0, kc[2], s1);
        s1 = wmma_bf16(qf1, kc[3], s1);

        // Prefetch next block's K fragments (land during softmax + PV).
        const bool more = (kb + 32) <= kb_end;
        v16bf kn[4];
        if (more) {
            kn[0] = load_frag_rows(Kh + (kb + 32) * HDIM +  0, HDIM);
            kn[1] = load_frag_rows(Kh + (kb + 32) * HDIM + 32, HDIM);
            kn[2] = load_frag_rows(Kh + (kb + 48) * HDIM +  0, HDIM);
            kn[3] = load_frag_rows(Kh + (kb + 48) * HDIM + 32, HDIM);
        }

        s0 = s0 * SCALE;
        s1 = s1 * SCALE;

        if (kb + 31 > t0) {                  // causal mask on diagonal blocks
#pragma unroll
            for (int e = 0; e < 8; ++e) {
                const int row_t = t0 + e + hi8;
                if (kb + cl      > row_t) s0[e] = -1.0e30f;
                if (kb + 16 + cl > row_t) s1[e] = -1.0e30f;
            }
        }

        // Online softmax (row lives across a 16-lane half).
#pragma unroll
        for (int e = 0; e < 8; ++e) {
            float a0 = s0[e], a1 = s1[e];
            float mx = fmaxf(a0, a1);
            mx = fmaxf(mx, __shfl_xor(mx, 1));
            mx = fmaxf(mx, __shfl_xor(mx, 2));
            mx = fmaxf(mx, __shfl_xor(mx, 4));
            mx = fmaxf(mx, __shfl_xor(mx, 8));
            const float mnew  = fmaxf(m_s[e], mx);
            const float alpha = __builtin_amdgcn_exp2f((m_s[e] - mnew) * L2E);
            const float p0 = __builtin_amdgcn_exp2f((a0 - mnew) * L2E);
            const float p1 = __builtin_amdgcn_exp2f((a1 - mnew) * L2E);
            float ps = p0 + p1;
            ps += __shfl_xor(ps, 1);
            ps += __shfl_xor(ps, 2);
            ps += __shfl_xor(ps, 4);
            ps += __shfl_xor(ps, 8);
            l_s[e] = l_s[e] * alpha + ps;
            m_s[e] = mnew;
#pragma unroll
            for (int c = 0; c < 4; ++c) o[c][e] *= alpha;
            Pl[(e + hi8) * 32 + cl]      = (__bf16)p0;
            Pl[(e + hi8) * 32 + cl + 16] = (__bf16)p1;
        }

        // Reload P as a 16x32 A-fragment (LDS ops in-order within wave).
        v16bf pf;
        {
            FragBF u;
            const __bf16* p = &Pl[(lane & 15) * 32 + hi8];
            u.q[0] = *(const v4u*)(p);
            u.q[1] = *(const v4u*)(p + 16);
            pf = u.f;
        }

        // O += P * V : V^T rows are head-dims, contiguous over keys.
#pragma unroll
        for (int c = 0; c < 4; ++c)
            o[c] = wmma_bf16(pf, vf[c], o[c]);

        if (more) {
#pragma unroll
            for (int i = 0; i < 4; ++i) kc[i] = kn[i];
        }
    }

#pragma unroll
    for (int e = 0; e < 8; ++e) {
        const float rl = 1.0f / l_s[e];
        const int t = t0 + e + hi8;
#pragma unroll
        for (int c = 0; c < 4; ++c) {
            const int d = c * 16 + cl;
            Ob[((size_t)(b * SEQ + t)) * DIMC + h * HDIM + d] =
                (__bf16)(o[c][e] * rl);
        }
    }
}

// ---------------------------------------------------------------- launch
extern "C" void kernel_launch(void* const* d_in, const int* in_sizes, int n_in,
                              void* d_out, int out_size, void* d_ws, size_t ws_size,
                              hipStream_t stream) {
    const float* x  = (const float*)d_in[0];
    const float* Wq = (const float*)d_in[1];
    const float* bq = (const float*)d_in[2];
    const float* Wk = (const float*)d_in[3];
    const float* bk = (const float*)d_in[4];
    const float* Wv = (const float*)d_in[5];
    const float* bv = (const float*)d_in[6];
    const float* Wo = (const float*)d_in[7];
    const float* bo = (const float*)d_in[8];

    char* ws = (char*)d_ws;
    const size_t SZ_X = (size_t)ROWS * DIMC * sizeof(__bf16);   // 8 MB
    const size_t SZ_W = (size_t)DIMC * DIMC * sizeof(__bf16);   // 2 MB
    __bf16* xb  = (__bf16*)(ws);
    __bf16* wqb = (__bf16*)(ws + SZ_X);
    __bf16* wkb = (__bf16*)(ws + SZ_X + 1 * SZ_W);
    __bf16* wvb = (__bf16*)(ws + SZ_X + 2 * SZ_W);
    __bf16* wob = (__bf16*)(ws + SZ_X + 3 * SZ_W);
    __bf16* Qb  = (__bf16*)(ws + SZ_X + 4 * SZ_W);
    __bf16* Kb  = (__bf16*)(ws + 2 * SZ_X + 4 * SZ_W);
    __bf16* Vt  = (__bf16*)(ws + 3 * SZ_X + 4 * SZ_W);
    __bf16* Ob  = (__bf16*)(ws + 4 * SZ_X + 4 * SZ_W);

    const int nx4 = ROWS * DIMC / 4;
    const int nw4 = DIMC * DIMC / 4;
    cvt_f32_bf16_kernel<<<(nx4 + 255) / 256, 256, 0, stream>>>((const v4f*)x,  (v4bf*)xb,  nx4);
    cvt_f32_bf16_kernel<<<(nw4 + 255) / 256, 256, 0, stream>>>((const v4f*)Wq, (v4bf*)wqb, nw4);
    cvt_f32_bf16_kernel<<<(nw4 + 255) / 256, 256, 0, stream>>>((const v4f*)Wk, (v4bf*)wkb, nw4);
    cvt_f32_bf16_kernel<<<(nw4 + 255) / 256, 256, 0, stream>>>((const v4f*)Wv, (v4bf*)wvb, nw4);
    cvt_f32_bf16_kernel<<<(nw4 + 255) / 256, 256, 0, stream>>>((const v4f*)Wo, (v4bf*)wob, nw4);

    dim3 ggrid(DIMC / 128, ROWS / 64);   // 8 x 64 workgroups of 4 waves
    dim3 gblk(128);
    gemm_bf16_kernel<GMODE_QK><<<ggrid, gblk, 0, stream>>>(xb, wqb, bq, Qb);
    gemm_bf16_kernel<GMODE_QK><<<ggrid, gblk, 0, stream>>>(xb, wkb, bk, Kb);
    gemm_bf16_kernel<GMODE_VT><<<ggrid, gblk, 0, stream>>>(xb, wvb, bv, Vt);

    dim3 agrid(SEQ / 16, BATCH * NHEAD);
    attention_kernel<<<agrid, dim3(32), 0, stream>>>(Qb, Kb, Vt, Ob);

    gemm_bf16_kernel<GMODE_F32><<<ggrid, gblk, 0, stream>>>(Ob, wob, bo, d_out);
}
